// BEVTrajSceneContextEncoder_83528523973292
// MI455X (gfx1250) — matrix operand order repaired and checked by reading, not compiled, not run on hardware
//
#include <hip/hip_runtime.h>
#include <hip/hip_bf16.h>

// ---------------------------------------------------------------------------
// BEVTrajSceneContextEncoder for MI455X (gfx1250, wave32, WMMA + TDM)
// B=16, N=1024, D=256, K=16, L=3, H=8, hd=32
// ---------------------------------------------------------------------------

#define BB 16
#define NN 1024
#define DD 256
#define KK 16
#define LL 3
#define HH 8
#define HD 32
#define FF 1024            // 4*D

typedef _Float16     v16h __attribute__((ext_vector_type(16)));
typedef _Float16     h8   __attribute__((ext_vector_type(8)));
typedef _Float16     h4   __attribute__((ext_vector_type(4)));
typedef float        v8f  __attribute__((ext_vector_type(8)));
typedef unsigned int u32x4 __attribute__((ext_vector_type(4)));
typedef int          i32x4 __attribute__((ext_vector_type(4)));
typedef int          i32x8 __attribute__((ext_vector_type(8)));

#if defined(__has_builtin)
#  if __has_builtin(__builtin_amdgcn_tensor_load_to_lds)
#    define HAS_TDM 1
#  endif
#endif
#ifndef HAS_TDM
#  define HAS_TDM 0
#endif

// ---------------------------------------------------------------------------
// TDM: 2D f16 tile (tile_dim0 = 32 halves wide) from global -> LDS, with
// LDS padding so each 64B row is followed by 16B pad (stride 80B = 40 halves).
// Issue from ONE wave only (EXEC is ignored by tensor ops; the branch below
// is wave-uniform). Completion via s_wait_tensorcnt.
// ---------------------------------------------------------------------------
#if HAS_TDM
__device__ __forceinline__ void tdm_load_tile_f16(unsigned lds_off, const void* gsrc,
                                                  int kd /*tensor width & stride, halves*/,
                                                  int rows /*tile_dim1*/)
{
    const unsigned long long ga = (unsigned long long)gsrc;
    u32x4 g0;
    g0[0] = 1u;                                           // count=1, user mode
    g0[1] = lds_off;                                      // lds_addr (bytes)
    g0[2] = (unsigned)ga;                                 // global_addr[31:0]
    g0[3] = (unsigned)((ga >> 32) & 0x01FFFFFFu) | (2u << 30);   // addr[56:32], type=2
    i32x8 g1;
    // data_size=1 (2B) | pad_enable | pad_interval=3 (16 DW = 64B) | pad_amount=3 (4 DW = 16B)
    g1[0] = (1 << 16) | (1 << 20) | (3 << 22) | (3 << 25);
    g1[1] = (kd & 0xFFFF) << 16;                          // tensor_dim0[15:0]  (bits 63:48)
    g1[2] = ((kd >> 16) & 0xFFFF) | (0x4000 << 16);       // tensor_dim0[31:16] | tensor_dim1 lo (big)
    g1[3] = 0 | (32 << 16);                               // tensor_dim1 hi | tile_dim0 = 32 halves
    g1[4] = rows & 0xFFFF;                                // tile_dim1 ; tile_dim2 = 0
    g1[5] = kd;                                           // tensor_dim0_stride[31:0] (halves)
    g1[6] = 0;                                            // stride[47:32] | dim1_stride lo
    g1[7] = 0;
    const i32x4 z4 = {0, 0, 0, 0};
#if defined(__clang_major__) && (__clang_major__ >= 23)
    const i32x8 z8 = {0, 0, 0, 0, 0, 0, 0, 0};
    __builtin_amdgcn_tensor_load_to_lds(g0, g1, z4, z4, z8, 0);
#else
    __builtin_amdgcn_tensor_load_to_lds(g0, g1, z4, z4, 0);
#endif
}
#endif

// ---------------------------------------------------------------------------
// Weight prep: f32 [mats][Kd][Nout] -> f16 transposed [mats][Nout][Kd]
// ---------------------------------------------------------------------------
__global__ void __launch_bounds__(256)
wtrans_kernel(const float* __restrict__ W, _Float16* __restrict__ Wt,
              int noutShift, int kShift)
{
    const int g   = blockIdx.x * 256 + threadIdx.x;
    const int mat = g >> (noutShift + kShift);
    const int r   = g & ((1 << (noutShift + kShift)) - 1);
    const int k   = r >> noutShift;
    const int n   = r & ((1 << noutShift) - 1);
    Wt[((size_t)mat << (noutShift + kShift)) + ((size_t)n << kShift) + k] = (_Float16)W[g];
}

// ---------------------------------------------------------------------------
// KNN: brute-force top-16 nearest neighbors over masked keys.
// ---------------------------------------------------------------------------
__global__ void __launch_bounds__(256)
knn_kernel(const float* __restrict__ xpos, const unsigned char* __restrict__ xmask,
           int* __restrict__ idx, float* __restrict__ nval)
{
    __shared__ float sx[NN], sy[NN], sz[NN], sm[NN];
    const int b     = blockIdx.x >> 2;
    const int chunk = blockIdx.x & 3;
    const int tid   = threadIdx.x;

    for (int j = tid; j < NN; j += 256) {
        const float* p = xpos + ((size_t)b * NN + j) * 3;
        sx[j] = p[0]; sy[j] = p[1]; sz[j] = p[2];
        sm[j] = xmask[b * NN + j] ? 0.0f : 1e30f;
    }
    __syncthreads();

    const int qn = chunk * 256 + tid;
    const float qx = sx[qn], qy = sy[qn], qz = sz[qn];

    float bd[KK]; int bi[KK];
#pragma unroll
    for (int t = 0; t < KK; ++t) { bd[t] = 3.0e38f; bi[t] = 0; }

    for (int j = 0; j < NN; ++j) {
        const float dx = sx[j] - qx, dy = sy[j] - qy, dz = sz[j] - qz;
        const float d2 = dx*dx + dy*dy + dz*dz + sm[j];
        if (d2 < bd[KK - 1]) {
            float cd = d2; int ci = j;
#pragma unroll
            for (int t = 0; t < KK; ++t) {
                if (cd < bd[t]) {
                    const float td = bd[t]; const int ti = bi[t];
                    bd[t] = cd; bi[t] = ci; cd = td; ci = ti;
                }
            }
        }
    }
    const int base = (b * NN + qn) * KK;
#pragma unroll
    for (int t = 0; t < KK; ++t) {
        idx[base + t]  = bi[t];
        nval[base + t] = (sm[bi[t]] == 0.0f) ? 1.0f : 0.0f;
    }
}

// ---------------------------------------------------------------------------
// Sine positional embedding on (x, y).
// ---------------------------------------------------------------------------
__global__ void __launch_bounds__(256)
sinembed_kernel(const float* __restrict__ xpos, float* __restrict__ pos)
{
    const int g = blockIdx.x * 256 + threadIdx.x;    // B*N*128 threads
    const int t = g & 127;
    const int n = (g >> 7) & (NN - 1);
    const int b = g >> 17;
    const float* p = xpos + ((size_t)b * NN + n) * 3;
    const float val  = (t < 64) ? p[1] : p[0];
    const int   tt   = t & 63;
    const float freq = powf(10000.0f, (float)tt * (1.0f / 64.0f));
    const float ph   = val * 6.283185307179586f / freq;
    float* o = pos + ((size_t)b * NN + n) * DD + (t < 64 ? 0 : 128) + tt * 2;
    o[0] = sinf(ph);
    o[1] = cosf(ph);
}

// ---------------------------------------------------------------------------
// WMMA GEMM:  Y[M,Nout] = act( (X[M,Kd] (+P)) @ Wt^T + bias )
// Wt is pre-transposed f16 [Nout][Kd]. Block tile 64x128, 8 waves as 4x2,
// each wave 16x64 (4 WMMAs reusing one A fragment). B tile (and A tile when
// IN_HALF) arrives via TDM tensor_load_to_lds with hardware LDS padding.
// ---------------------------------------------------------------------------
template<bool IN_HALF, bool ADD_P, bool OUT_HALF, bool RELU>
__global__ void __launch_bounds__(256)
gemm_tile(const void* __restrict__ Xv, const float* __restrict__ P,
          const _Float16* __restrict__ Wt, const float* __restrict__ bias,
          void* __restrict__ Yv, int Kd, int Nout)
{
    __shared__ _Float16 As[64][40];    // 64 rows x 32 (+8 pad) halves
    __shared__ _Float16 Bs[128][40];   // 128 cols (n-major) x 32 (+8 pad) halves

    const int tid     = threadIdx.x;
    const int rowBase = blockIdx.y * 64;
    const int colBase = blockIdx.x * 128;
    const int wave = tid >> 5, lane = tid & 31;
    const int wm = wave >> 1, wn = wave & 1;        // 4 x 2 wave grid
    const int hi = lane >> 4, lr = lane & 15;

    const float*    Xf = (const float*)Xv;
    const _Float16* Xh = (const _Float16*)Xv;

#if HAS_TDM
    const unsigned asOff = (unsigned)(size_t)&As[0][0];
    const unsigned bsOff = (unsigned)(size_t)&Bs[0][0];
#endif

    v8f c[4] = {};

    for (int k0 = 0; k0 < Kd; k0 += 32) {
#if HAS_TDM
        if (tid < 32) {   // wave-uniform: only wave 0 issues TDM descriptors
            tdm_load_tile_f16(bsOff, Wt + (size_t)colBase * Kd + k0, Kd, 128);
            if constexpr (IN_HALF)
                tdm_load_tile_f16(asOff, Xh + (size_t)rowBase * Kd + k0, Kd, 64);
        }
#else
        {   // manual B tile: 128 rows x 32 halves
            const int row = tid >> 1, c8 = (tid & 1) << 4;
            const h8 w0 = *(const h8*)&Wt[(size_t)(colBase + row) * Kd + k0 + c8];
            const h8 w1 = *(const h8*)&Wt[(size_t)(colBase + row) * Kd + k0 + c8 + 8];
            *(h8*)&Bs[row][c8]     = w0;
            *(h8*)&Bs[row][c8 + 8] = w1;
        }
        if constexpr (IN_HALF) {
            const int r  = tid >> 2, c8 = (tid & 3) << 3;
            const h8 x = *(const h8*)&Xh[(size_t)(rowBase + r) * Kd + k0 + c8];
            *(h8*)&As[r][c8] = x;
        }
#endif
        if constexpr (!IN_HALF) {   // A tile: f32 (+pos) -> f16
#pragma unroll
            for (int it = 0; it < 2; ++it) {
                const int f  = tid + it * 256;       // 512 float4 chunks
                const int r  = f >> 3;
                const int c4 = (f & 7) << 2;
                float4 x = *(const float4*)&Xf[(size_t)(rowBase + r) * Kd + k0 + c4];
                if constexpr (ADD_P) {
                    const float4 pv = *(const float4*)&P[(size_t)(rowBase + r) * Kd + k0 + c4];
                    x.x += pv.x; x.y += pv.y; x.z += pv.z; x.w += pv.w;
                }
                h4 hv = { (_Float16)x.x, (_Float16)x.y, (_Float16)x.z, (_Float16)x.w };
                *(h4*)&As[r][c4] = hv;
            }
            if (k0 + 32 < Kd)       // global_prefetch_b8 for next K-tile
                __builtin_prefetch(&Xf[(size_t)(rowBase + (tid >> 3)) * Kd + k0 + 32], 0, 1);
        }
#if HAS_TDM && __has_builtin(__builtin_amdgcn_s_wait_tensorcnt)
        __builtin_amdgcn_s_wait_tensorcnt(0);   // waves 1..7 have count 0 -> free
#endif
        __syncthreads();

        // ------- fragments (ISA 7.12.2) -------
        v16h a;
        {   // A: row = wm*16+lr; K runs {kb..kb+7, kb+16..kb+23}, kb = hi*8
            const int r = wm * 16 + lr, kb = hi * 8;
            const h8 lo = *(const h8*)&As[r][kb];
            const h8 hv = *(const h8*)&As[r][kb + 16];
#pragma unroll
            for (int i = 0; i < 8; ++i) { a[i] = lo[i]; a[i + 8] = hv[i]; }
        }
#pragma unroll
        for (int nsub = 0; nsub < 4; ++nsub) {
            const int nrow = wn * 64 + nsub * 16 + lr;   // B: K runs {hi*8..+7, hi*8+16..+23}, same as A
            const h8 lo = *(const h8*)&Bs[nrow][hi * 8];
            const h8 hv = *(const h8*)&Bs[nrow][hi * 8 + 16];
            v16h b;
#pragma unroll
            for (int i = 0; i < 8; ++i) { b[i] = lo[i]; b[i + 8] = hv[i]; }
            c[nsub] = __builtin_amdgcn_wmma_f32_16x16x32_f16(false, a, false, b,
                                                             (short)0, c[nsub], false, false);
        }
        __syncthreads();
    }

    // ------- epilogue: bias (+relu), store f32 or f16 -------
#pragma unroll
    for (int nsub = 0; nsub < 4; ++nsub) {
        const int n  = colBase + wn * 64 + nsub * 16 + lr;
        const float bv = bias[n];
#pragma unroll
        for (int r = 0; r < 8; ++r) {
            const int m = rowBase + wm * 16 + hi * 8 + r;   // D-layout: M = vgpr + 8*(lane>=16)
            float val = c[nsub][r] + bv;
            if constexpr (RELU) val = fmaxf(val, 0.0f);
            if constexpr (OUT_HALF)
                ((_Float16*)Yv)[(size_t)m * Nout + n] = (_Float16)val;
            else
                ((float*)Yv)[(size_t)m * Nout + n] = val;
        }
    }
}

// ---------------------------------------------------------------------------
// Local attention over K=16 neighbors. One thread per (b, n, head).
// ---------------------------------------------------------------------------
__global__ void __launch_bounds__(256)
attn_kernel(const float* __restrict__ q, const float* __restrict__ k,
            const float* __restrict__ v, const int* __restrict__ idx,
            const float* __restrict__ nval, float* __restrict__ o)
{
    const int t  = blockIdx.x * 256 + threadIdx.x;   // B*N*H threads
    const int hh = t & (HH - 1);
    const int n  = (t >> 3) & (NN - 1);
    const int b  = t >> 13;

    const float* qp = q + ((size_t)b * NN + n) * DD + hh * HD;
    float qv[HD];
#pragma unroll
    for (int d = 0; d < HD; ++d) qv[d] = qp[d];

    const int*   ip = idx  + ((size_t)b * NN + n) * KK;
    const float* vp = nval + ((size_t)b * NN + n) * KK;
    int nj[KK]; float lg[KK];
    float mx = -3.0e38f;
#pragma unroll
    for (int j = 0; j < KK; ++j) {
        nj[j] = ip[j];
        const float* kp = k + ((size_t)b * NN + nj[j]) * DD + hh * HD;
        float dot = 0.0f;
#pragma unroll
        for (int d = 0; d < HD; ++d) dot += qv[d] * kp[d];
        float l = dot * 0.17677669529663687f;        // 1/sqrt(32)
        l = (vp[j] != 0.0f) ? l : -1.0e9f;
        lg[j] = l;
        mx = fmaxf(mx, l);
    }
    float se = 0.0f;
#pragma unroll
    for (int j = 0; j < KK; ++j) { lg[j] = __expf(lg[j] - mx); se += lg[j]; }
    const float inv = 1.0f / se;

    float acc[HD];
#pragma unroll
    for (int d = 0; d < HD; ++d) acc[d] = 0.0f;
#pragma unroll
    for (int j = 0; j < KK; ++j) {
        const float w = lg[j] * inv;
        const float* vvp = v + ((size_t)b * NN + nj[j]) * DD + hh * HD;
#pragma unroll
        for (int d = 0; d < HD; ++d) acc[d] += w * vvp[d];
    }
    float* op = o + ((size_t)b * NN + n) * DD + hh * HD;
#pragma unroll
    for (int d = 0; d < HD; ++d) op[d] = acc[d];
}

// ---------------------------------------------------------------------------
// out = LayerNorm(a + b) * g + be    (one wave32 per 256-wide row)
// ---------------------------------------------------------------------------
__global__ void __launch_bounds__(256)
add_ln_kernel(const float* __restrict__ a, const float* __restrict__ bsrc,
              const float* __restrict__ g, const float* __restrict__ be,
              float* __restrict__ out)
{
    const int row  = blockIdx.x * 8 + (threadIdx.x >> 5);
    const int lane = threadIdx.x & 31;
    const float* pa = a    + (size_t)row * DD;
    const float* pb = bsrc + (size_t)row * DD;

    float vals[8];
    float s = 0.0f;
#pragma unroll
    for (int i = 0; i < 8; ++i) {
        vals[i] = pa[lane + i * 32] + pb[lane + i * 32];
        s += vals[i];
    }
#pragma unroll
    for (int off = 16; off > 0; off >>= 1) s += __shfl_xor(s, off, 32);
    const float mu = s * (1.0f / DD);

    float vs = 0.0f;
#pragma unroll
    for (int i = 0; i < 8; ++i) { const float d = vals[i] - mu; vs += d * d; }
#pragma unroll
    for (int off = 16; off > 0; off >>= 1) vs += __shfl_xor(vs, off, 32);
    const float rstd = rsqrtf(vs * (1.0f / DD) + 1e-5f);

    float* po = out + (size_t)row * DD;
#pragma unroll
    for (int i = 0; i < 8; ++i) {
        const int ccol = lane + i * 32;
        po[ccol] = (vals[i] - mu) * rstd * g[ccol] + be[ccol];
    }
}

// ---------------------------------------------------------------------------
// Final scatter-back: invalid tokens -> 0
// ---------------------------------------------------------------------------
__global__ void __launch_bounds__(256)
maskout_kernel(const float* __restrict__ h, const unsigned char* __restrict__ mask,
               float* __restrict__ out)
{
    const int g  = blockIdx.x * 256 + threadIdx.x;   // B*N*D threads
    const int bn = g >> 8;                           // D = 256
    out[g] = mask[bn] ? h[g] : 0.0f;
}

// ---------------------------------------------------------------------------
// Orchestration
// ---------------------------------------------------------------------------
extern "C" void kernel_launch(void* const* d_in, const int* in_sizes, int n_in,
                              void* d_out, int out_size, void* d_ws, size_t ws_size,
                              hipStream_t stream)
{
    const float*         x     = (const float*)d_in[0];
    const unsigned char* xmask = (const unsigned char*)d_in[1];   // jax bool -> 1 byte
    const float*         xpos  = (const float*)d_in[2];
    const float *Wq = (const float*)d_in[3],  *bq = (const float*)d_in[4];
    const float *Wk = (const float*)d_in[5],  *bk = (const float*)d_in[6];
    const float *Wv = (const float*)d_in[7],  *bv = (const float*)d_in[8];
    const float *Wo = (const float*)d_in[9],  *bo = (const float*)d_in[10];
    const float *W1 = (const float*)d_in[11], *b1 = (const float*)d_in[12];
    const float *W2 = (const float*)d_in[13], *b2 = (const float*)d_in[14];
    const float *g1 = (const float*)d_in[15], *be1 = (const float*)d_in[16];
    const float *g2 = (const float*)d_in[17], *be2 = (const float*)d_in[18];

    const size_t BN  = (size_t)BB * NN;          // 16384
    const size_t BND = BN * DD;                  // 4,194,304

    char* ws = (char*)d_ws;
    size_t off = 0;
    auto take = [&](size_t bytes) -> void* {
        void* p = ws + off;
        off += (bytes + 255) & ~(size_t)255;
        return p;
    };
    float*     pos  = (float*)take(BND * 4);
    float*     h0   = (float*)take(BND * 4);
    float*     h1   = (float*)take(BND * 4);
    float*     qb   = (float*)take(BND * 4);
    float*     kbuf = (float*)take(BND * 4);
    float*     vbuf = (float*)take(BND * 4);
    float*     ob   = (float*)take(BND * 4);
    _Float16*  ff1h = (_Float16*)take(BN * FF * 2);
    int*       idxb = (int*)take(BN * KK * 4);
    float*     nvb  = (float*)take(BN * KK * 4);
    _Float16*  wqt  = (_Float16*)take((size_t)LL * DD * DD * 2);
    _Float16*  wkt  = (_Float16*)take((size_t)LL * DD * DD * 2);
    _Float16*  wvt  = (_Float16*)take((size_t)LL * DD * DD * 2);
    _Float16*  wot  = (_Float16*)take((size_t)LL * DD * DD * 2);
    _Float16*  w1t  = (_Float16*)take((size_t)LL * DD * FF * 2);
    _Float16*  w2t  = (_Float16*)take((size_t)LL * FF * DD * 2);
    (void)ws_size;

    // h = x
    hipMemcpyAsync(h0, x, BND * 4, hipMemcpyDeviceToDevice, stream);

    // weight prep: convert to f16 + transpose to [Nout][Kd]
    wtrans_kernel<<<(LL * DD * DD) / 256, 256, 0, stream>>>(Wq, wqt, 8, 8);
    wtrans_kernel<<<(LL * DD * DD) / 256, 256, 0, stream>>>(Wk, wkt, 8, 8);
    wtrans_kernel<<<(LL * DD * DD) / 256, 256, 0, stream>>>(Wv, wvt, 8, 8);
    wtrans_kernel<<<(LL * DD * DD) / 256, 256, 0, stream>>>(Wo, wot, 8, 8);
    wtrans_kernel<<<(LL * DD * FF) / 256, 256, 0, stream>>>(W1, w1t, 10, 8);
    wtrans_kernel<<<(LL * FF * DD) / 256, 256, 0, stream>>>(W2, w2t, 8, 10);

    // KNN + positional embedding (layer-invariant)
    knn_kernel<<<BB * (NN / 256), 256, 0, stream>>>(xpos, xmask, idxb, nvb);
    sinembed_kernel<<<(BB * NN * 128) / 256, 256, 0, stream>>>(xpos, pos);

    const dim3 gridD(DD / 128, (int)(BN / 64));   // (2, 256)  Nout=256
    const dim3 gridF(FF / 128, (int)(BN / 64));   // (8, 256)  Nout=1024

    for (int l = 0; l < LL; ++l) {
        const _Float16 *wqt_l = wqt + (size_t)l * DD * DD;
        const _Float16 *wkt_l = wkt + (size_t)l * DD * DD;
        const _Float16 *wvt_l = wvt + (size_t)l * DD * DD;
        const _Float16 *wot_l = wot + (size_t)l * DD * DD;
        const _Float16 *w1t_l = w1t + (size_t)l * DD * FF;
        const _Float16 *w2t_l = w2t + (size_t)l * FF * DD;
        const float *bq_l = bq + (size_t)l * DD, *bk_l = bk + (size_t)l * DD;
        const float *bv_l = bv + (size_t)l * DD, *bo_l = bo + (size_t)l * DD;
        const float *b1_l = b1 + (size_t)l * FF, *b2_l = b2 + (size_t)l * DD;

        // q = (h+pos)@Wq+bq ; k = (h+pos)@Wk+bk ; v = h@Wv+bv
        gemm_tile<false, true,  false, false><<<gridD, 256, 0, stream>>>(h0, pos, wqt_l, bq_l, qb,   DD, DD);
        gemm_tile<false, true,  false, false><<<gridD, 256, 0, stream>>>(h0, pos, wkt_l, bk_l, kbuf, DD, DD);
        gemm_tile<false, false, false, false><<<gridD, 256, 0, stream>>>(h0, nullptr, wvt_l, bv_l, vbuf, DD, DD);

        // local attention over 16 neighbors
        attn_kernel<<<(BB * NN * HH) / 256, 256, 0, stream>>>(qb, kbuf, vbuf, idxb, nvb, ob);

        // o @ Wo + bo  -> qb (free)
        gemm_tile<false, false, false, false><<<gridD, 256, 0, stream>>>(ob, nullptr, wot_l, bo_l, qb, DD, DD);

        // h1 = LN(h0 + proj)
        add_ln_kernel<<<(int)(BN / 8), 256, 0, stream>>>(h0, qb, g1 + (size_t)l * DD, be1 + (size_t)l * DD, h1);

        // ff1 = relu(h1 @ W1 + b1)  (f16 output feeds next GEMM directly)
        gemm_tile<false, false, true, true><<<gridF, 256, 0, stream>>>(h1, nullptr, w1t_l, b1_l, ff1h, DD, FF);

        // ff2 = ff1 @ W2 + b2 -> kbuf (free), A tile also via TDM (f16 input)
        gemm_tile<true, false, false, false><<<gridD, 256, 0, stream>>>(ff1h, nullptr, w2t_l, b2_l, kbuf, FF, DD);

        // h0 = LN(h1 + ff2)
        add_ln_kernel<<<(int)(BN / 8), 256, 0, stream>>>(h1, kbuf, g2 + (size_t)l * DD, be2 + (size_t)l * DD, h0);
    }

    // zero invalid tokens
    maskout_kernel<<<(int)(BND / 256), 256, 0, stream>>>(h0, xmask, (float*)d_out);
}
